// SimpleGRU_56745107914837
// MI455X (gfx1250) — matrix operand order repaired and checked by reading, not compiled, not run on hardware
//
#include <hip/hip_runtime.h>

#define B_SZ 512
#define T_SZ 512
#define I_SZ 32
#define H_SZ 64
#define G_SZ 192      // 3*H gate columns
#define NWAVE 12      // waves doing WMMA (12 x 16 = 192 gate columns)
#define NTHR  512     // 16 waves; combine/init loops become uniform (2 iters)
#define SH_STR 80     // padded bf16 row stride (160B, 16B-aligned) for hidden tiles
#define G_STR 196     // padded f32 row stride for gate staging buffers

typedef __attribute__((ext_vector_type(16))) __bf16 v16bf;
typedef __attribute__((ext_vector_type(8)))  __bf16 v8bf;
typedef __attribute__((ext_vector_type(8)))  float  v8f;

// A-operand (16x32 bf16) from an LDS hidden-state row.
// Lane L: row = L&15, hv = L>>4; elements K = kk+hv*8+i (i<8) and kk+16+hv*8+i.
__device__ __forceinline__ v16bf a_from_lds(const __bf16* row, int kk, int hv) {
  const v8bf lo = *(const v8bf*)(row + kk + hv * 8);
  const v8bf hi = *(const v8bf*)(row + kk + 16 + hv * 8);
  v16bf a;
#pragma unroll
  for (int i = 0; i < 8; ++i) { a[i] = lo[i]; a[8 + i] = hi[i]; }
  return a;
}

// A-operand (16x32) straight from global f32 (x_t tile), converted to bf16.
__device__ __forceinline__ v16bf a_from_global32(const float* p, int hv) {
  v16bf a;
#pragma unroll
  for (int i = 0; i < 8; ++i) a[i] = (__bf16)p[hv * 8 + i];
#pragma unroll
  for (int i = 0; i < 8; ++i) a[8 + i] = (__bf16)p[16 + hv * 8 + i];
  return a;
}

// B-operand (32x16): lane holds column n = ncol, K = kk+hv*8+i (i<8) and kk+16+hv*8+i.
// W is row-major [192][ldw]; B[k][n] = W[n][k], so fixed-n K-run is contiguous.
__device__ __forceinline__ v16bf b_from_wt(const float* W, int ldw, int n, int kk, int hv) {
  const float* p = W + n * ldw + kk;
  v16bf b;
#pragma unroll
  for (int i = 0; i < 8; ++i) { b[i] = (__bf16)p[hv * 8 + i]; b[8 + i] = (__bf16)p[16 + hv * 8 + i]; }
  return b;
}

__device__ __forceinline__ v8f bias_splat(float b) {
  v8f c;
#pragma unroll
  for (int r = 0; r < 8; ++r) c[r] = b;   // all 8 rows of a lane share one column
  return c;
}

__device__ __forceinline__ v8f wmma_bf16(v16bf a, v16bf b, v8f c) {
  return __builtin_amdgcn_wmma_f32_16x16x32_bf16(false, a, false, b, (short)0, c,
                                                 false, false);
}

// Fast sigmoid: single v_exp_f32 + single v_rcp_f32 (no IEEE-div Newton chain).
__device__ __forceinline__ float fast_sigmoid(float v) {
  return __builtin_amdgcn_rcpf(1.0f + __expf(-v));
}

// Fast tanh: native gfx1250 v_tanh_f32 if the builtin exists, else exp/rcp form.
__device__ __forceinline__ float fast_tanh(float v) {
#if __has_builtin(__builtin_amdgcn_tanhf)
  return __builtin_amdgcn_tanhf(v);
#else
  float ax = __builtin_fabsf(v);
  float e  = __expf(-2.0f * ax);                     // in (0,1], no overflow
  float r  = (1.0f - e) * __builtin_amdgcn_rcpf(1.0f + e);
  return __builtin_copysignf(r, v);
#endif
}

__global__ __launch_bounds__(NTHR)
void gru2_fused_wmma_kernel(const float* __restrict__ x,
                            const float* __restrict__ Wih0, const float* __restrict__ Whh0,
                            const float* __restrict__ bih0, const float* __restrict__ bhh0,
                            const float* __restrict__ Wih1, const float* __restrict__ Whh1,
                            const float* __restrict__ bih1, const float* __restrict__ bhh1,
                            const float* __restrict__ fcw,  const float* __restrict__ fcb,
                            float* __restrict__ out) {
  __shared__ alignas(16) float  s0f[16][H_SZ];     // layer-0 hidden, f32 master
  __shared__ alignas(16) float  s1f[16][H_SZ];     // layer-1 hidden, f32 master
  __shared__ alignas(16) __bf16 s0h[16][SH_STR];   // bf16 copies for WMMA A-operand
  __shared__ alignas(16) __bf16 s1h[16][SH_STR];
  __shared__ alignas(16) float  gbi[16][G_STR];    // input-path gate pre-acts  [16][192]
  __shared__ alignas(16) float  gbh[16][G_STR];    // hidden-path gate pre-acts [16][192]

  const int  tid   = threadIdx.x;
  const int  wave  = tid >> 5;
  const int  lane  = tid & 31;
  const int  ln16  = lane & 15;
  const int  hv    = lane >> 4;                    // lane half (WMMA layout)
  // Force the wave-id to an SGPR so the WMMA-phase guard becomes a scalar
  // branch (s_cmp + s_cbranch) instead of per-lane exec masking; this also
  // structurally guarantees EXEC==all-ones inside the WMMA region.
  const bool mm    = (__builtin_amdgcn_readfirstlane(wave) < NWAVE);
  const int  gwave = (wave < NWAVE) ? wave : 0;    // clamp so loads stay in-bounds
  const int  ncol  = (gwave << 4) + ln16;          // gate column 0..191
  const int  row0  = blockIdx.x * 16;              // batch rows [row0, row0+16)

  // init hidden states to zero: 1024 elements over 512 threads, uniform 2 iters
#pragma unroll
  for (int rep = 0; rep < 2; ++rep) {
    int idx = tid + rep * NTHR;
    int m = idx >> 6, c = idx & 63;
    s0f[m][c] = 0.f; s1f[m][c] = 0.f;
    s0h[m][c] = (__bf16)0.f; s1h[m][c] = (__bf16)0.f;
  }
  __syncthreads();

  // Per-wave weight column tiles live in registers for the whole scan (7 x 8 VGPRs).
  const v16bf wih0   = b_from_wt(Wih0, I_SZ, ncol, 0,  hv);
  const v16bf whh0_0 = b_from_wt(Whh0, H_SZ, ncol, 0,  hv);
  const v16bf whh0_1 = b_from_wt(Whh0, H_SZ, ncol, 32, hv);
  const v16bf wih1_0 = b_from_wt(Wih1, H_SZ, ncol, 0,  hv);
  const v16bf wih1_1 = b_from_wt(Wih1, H_SZ, ncol, 32, hv);
  const v16bf whh1_0 = b_from_wt(Whh1, H_SZ, ncol, 0,  hv);
  const v16bf whh1_1 = b_from_wt(Whh1, H_SZ, ncol, 32, hv);
  const v8f   cbi0 = bias_splat(bih0[ncol]);
  const v8f   cbh0 = bias_splat(bhh0[ncol]);
  const v8f   cbi1 = bias_splat(bih1[ncol]);
  const v8f   cbh1 = bias_splat(bhh1[ncol]);

  const float*  xrow  = x + (size_t)(row0 + ln16) * T_SZ * I_SZ;  // this lane's row
  const __bf16* a0row = &s0h[ln16][0];
  const __bf16* a1row = &s1h[ln16][0];

  for (int t = 0; t < T_SZ; ++t) {
    // ---------------- layer 0: gi = x_t@Wih0^T + bi0 ; gh = s0@Whh0^T + bh0 ----
    if (mm) {                                      // scalar-uniform branch
      const float* xp = xrow + t * I_SZ;
      __builtin_prefetch(xp + I_SZ, 0, 1);         // global_prefetch_b8, next step
      v8f ai = wmma_bf16(a_from_global32(xp, hv), wih0, cbi0);
      v8f ah = wmma_bf16(a_from_lds(a0row, 0,  hv), whh0_0, cbh0);
      ah     = wmma_bf16(a_from_lds(a0row, 32, hv), whh0_1, ah);
#pragma unroll
      for (int r = 0; r < 8; ++r) {                // D layout: (M = hv*8+r, N = ncol)
        gbi[hv * 8 + r][ncol] = ai[r];
        gbh[hv * 8 + r][ncol] = ah[r];
      }
    }
    __syncthreads();
#pragma unroll
    for (int rep = 0; rep < 2; ++rep) {            // uniform, branch-free combine
      int idx = tid + rep * NTHR;
      int m = idx >> 6, c = idx & 63;
      float r = fast_sigmoid(gbi[m][c] + gbh[m][c]);
      float z = fast_sigmoid(gbi[m][H_SZ + c] + gbh[m][H_SZ + c]);
      float n = fast_tanh(gbi[m][2 * H_SZ + c] + r * gbh[m][2 * H_SZ + c]);
      float hn = (1.f - z) * n + z * s0f[m][c];
      s0f[m][c] = hn;
      s0h[m][c] = (__bf16)hn;
    }
    __syncthreads();

    // ---------------- layer 1: gi = s0@Wih1^T + bi1 ; gh = s1@Whh1^T + bh1 ----
    if (mm) {
      v8f ai1 = wmma_bf16(a_from_lds(a0row, 0,  hv), wih1_0, cbi1);
      ai1     = wmma_bf16(a_from_lds(a0row, 32, hv), wih1_1, ai1);
      v8f ah1 = wmma_bf16(a_from_lds(a1row, 0,  hv), whh1_0, cbh1);
      ah1     = wmma_bf16(a_from_lds(a1row, 32, hv), whh1_1, ah1);
#pragma unroll
      for (int r = 0; r < 8; ++r) {
        gbi[hv * 8 + r][ncol] = ai1[r];
        gbh[hv * 8 + r][ncol] = ah1[r];
      }
    }
    __syncthreads();
#pragma unroll
    for (int rep = 0; rep < 2; ++rep) {
      int idx = tid + rep * NTHR;
      int m = idx >> 6, c = idx & 63;
      float r = fast_sigmoid(gbi[m][c] + gbh[m][c]);
      float z = fast_sigmoid(gbi[m][H_SZ + c] + gbh[m][H_SZ + c]);
      float n = fast_tanh(gbi[m][2 * H_SZ + c] + r * gbh[m][2 * H_SZ + c]);
      float hn = (1.f - z) * n + z * s1f[m][c];
      s1f[m][c] = hn;
      s1h[m][c] = (__bf16)hn;
    }
    __syncthreads();
  }

  // Final FC head: out[row] = dot(s1[row], fc_w) + fc_b  (16 rows, trivially VALU)
  if (tid < 16) {
    float acc = fcb[0];
#pragma unroll
    for (int k = 0; k < H_SZ; ++k) acc += s1f[tid][k] * fcw[k];
    out[row0 + tid] = acc;
  }
}

extern "C" void kernel_launch(void* const* d_in, const int* in_sizes, int n_in,
                              void* d_out, int out_size, void* d_ws, size_t ws_size,
                              hipStream_t stream) {
  (void)in_sizes; (void)n_in; (void)d_ws; (void)ws_size; (void)out_size;
  const float* x    = (const float*)d_in[0];
  const float* Wih0 = (const float*)d_in[1];
  const float* Whh0 = (const float*)d_in[2];
  const float* bih0 = (const float*)d_in[3];
  const float* bhh0 = (const float*)d_in[4];
  const float* Wih1 = (const float*)d_in[5];
  const float* Whh1 = (const float*)d_in[6];
  const float* bih1 = (const float*)d_in[7];
  const float* bhh1 = (const float*)d_in[8];
  const float* fcw  = (const float*)d_in[9];
  const float* fcb  = (const float*)d_in[10];

  gru2_fused_wmma_kernel<<<B_SZ / 16, NTHR, 0, stream>>>(
      x, Wih0, Whh0, bih0, bhh0, Wih1, Whh1, bih1, bhh1, fcw, fcb, (float*)d_out);
}
